// HierarchicalCodebookGrounding_57449482551998
// MI455X (gfx1250) — matrix operand order repaired and checked by reading, not compiled, not run on hardware
//
#include <hip/hip_runtime.h>
#include <hip/hip_bf16.h>
#include <stdint.h>

#define D_DIM   256
#define N_TOK   4096
#define N_BATCH 16
#define M_TOT   672       // 16 cat + 128 type + 512 var + 16 spatial
#define ATTN_WAVES 6
#define CHUNK   32

typedef __bf16 bf16_t;
typedef __bf16 v16bf __attribute__((ext_vector_type(16)));
typedef float  v8f   __attribute__((ext_vector_type(8)));
typedef unsigned int u32x4 __attribute__((ext_vector_type(4)));
typedef int          i32x4 __attribute__((ext_vector_type(4)));
typedef int          i32x8 __attribute__((ext_vector_type(8)));

__device__ inline v8f wmma_bf16(v16bf a, v16bf b, v8f c) {
  // (neg_a, A, neg_b, B, c_mod, C, reuse_a, reuse_b) -> v_wmma_f32_16x16x32_bf16
  return __builtin_amdgcn_wmma_f32_16x16x32_bf16(false, a, false, b, (short)0, c,
                                                 false, false);
}

// A-matrix fragment (16x32 bf16) from row-major src, tile at (row0, k0).
// Layout (ISA 7.12.2): lanes 0-15 row=lane, K 0..7 & 16..23; lanes 16-31 same row,
// K 8..15 & 24..31.
template <typename T>
__device__ inline v16bf load_A(const T* __restrict__ src, int ld, int row0, int k0) {
  const int lane = threadIdx.x & 31;
  const int h = lane >> 4;
  const T* p = src + (size_t)(row0 + (lane & 15)) * ld + k0 + h * 8;
  v16bf a;
#pragma unroll
  for (int i = 0; i < 8; ++i) {
    a[i]     = (bf16_t)(float)p[i];       // K = k0 + h*8 + i
    a[i + 8] = (bf16_t)(float)p[16 + i];  // K = k0 + 16 + h*8 + i
  }
  return a;
}

// B-matrix fragment (32x16 bf16) where B[k][n] = src[n0+n][k0+k] (src row-major).
// Layout: lanes 0-15 col=lane K 0..7 & 16..23; lanes 16-31 col=lane-16 K 8..15 & 24..31.
template <typename T>
__device__ inline v16bf load_Bt(const T* __restrict__ src, int ld, int n0, int k0) {
  const int lane = threadIdx.x & 31;
  const int h = lane >> 4;
  const T* p = src + (size_t)(n0 + (lane & 15)) * ld + k0 + h * 8;
  v16bf b;
#pragma unroll
  for (int i = 0; i < 8; ++i) {
    b[i]     = (bf16_t)(float)p[i];
    b[i + 8] = (bf16_t)(float)p[16 + i];
  }
  return b;
}

// TDM: DMA a 32x256 bf16 tile (contiguous rows, stride 256) from global into LDS,
// with hardware row padding of 4 DWORDs every 128 DWORDs -> LDS rows of 264 bf16.
// D# bitfields per CDNA5 ISA §8.3-8.6. Issued per-wave (EXEC ignored); caller must
// guard so only one wave issues, then s_wait_tensorcnt + barrier before LDS reads.
__device__ inline void tdm_load_k_tile(const bf16_t* gsrc, uint32_t lds_off) {
  const uint64_t ga = (uint64_t)(uintptr_t)gsrc;
  u32x4 g0;
  g0[0] = 1u;                                                 // count=1 (valid), user
  g0[1] = lds_off;                                            // lds_addr (bytes)
  g0[2] = (uint32_t)(ga & 0xFFFFFFFFu);                       // global_addr[31:0]
  g0[3] = (uint32_t)((ga >> 32) & 0x01FFFFFFu) | (2u << 30);  // addr[56:32] | type=2
  i32x8 g1;
  g1[0] = (1 << 16)   // data_size = 1 -> 2 bytes/elem
        | (1 << 20)   // pad_enable
        | (6 << 22)   // pad_interval: 128 DWORDs (one 256-elem bf16 row)
        | (3 << 25);  // pad_amount: 4 DWORDs (8 bf16) -> 264-elem LDS rows
  g1[1] = (D_DIM & 0xFFFF) << 16;   // tensor_dim0 = 256 (bits 79:48, low half)
  g1[2] = (CHUNK & 0xFFFF) << 16;   // tensor_dim0 hi=0 | tensor_dim1 = 32
  g1[3] = (D_DIM & 0xFFFF) << 16;   // tensor_dim1 hi=0 | tile_dim0 = 256
  g1[4] = CHUNK;                    // tile_dim1 = 32, tile_dim2 = 0
  g1[5] = D_DIM;                    // tensor_dim0_stride = 256 (48-bit, low word)
  g1[6] = 0;
  g1[7] = 0;
  i32x4 gz; gz[0] = 0; gz[1] = 0; gz[2] = 0; gz[3] = 0;
#if __clang_major__ >= 23
  i32x8 gz8;
  gz8[0] = 0; gz8[1] = 0; gz8[2] = 0; gz8[3] = 0;
  gz8[4] = 0; gz8[5] = 0; gz8[6] = 0; gz8[7] = 0;
  __builtin_amdgcn_tensor_load_to_lds(g0, g1, gz, gz, gz8, 0);
#else
  __builtin_amdgcn_tensor_load_to_lds(g0, g1, gz, gz, 0);
#endif
}

// ---------------------------------------------------------------- cvt: X -> bf16
__global__ __launch_bounds__(256) void hcg_cvt_kernel(const float* __restrict__ x,
                                                      bf16_t* __restrict__ y, int n4) {
  int i = blockIdx.x * 256 + threadIdx.x;
  if (i < n4) {
    const float4 v = ((const float4*)x)[i];
    union { bf16_t h[4]; uint64_t u; } o;
    o.h[0] = (bf16_t)v.x; o.h[1] = (bf16_t)v.y;
    o.h[2] = (bf16_t)v.z; o.h[3] = (bf16_t)v.w;
    ((uint64_t*)y)[i] = o.u;
  }
}

// ------------------------------------------------- kproj: K = X @ k_w^T + k_b (bf16)
__global__ __launch_bounds__(256) void hcg_kproj_kernel(const float* __restrict__ X,
                                                        const float* __restrict__ kw,
                                                        const float* __restrict__ kb,
                                                        bf16_t* __restrict__ Kb) {
  const int wid  = blockIdx.x * 8 + (threadIdx.x >> 5);
  const int row0 = (wid >> 4) * 16;     // token tile
  const int col0 = (wid & 15) * 16;     // output-dim tile
  v8f c;
#pragma unroll
  for (int r = 0; r < 8; ++r) c[r] = 0.0f;
#pragma unroll
  for (int kk = 0; kk < 8; ++kk) {
    v16bf a  = load_A(X, D_DIM, row0, kk * 32);
    v16bf bb = load_Bt(kw, D_DIM, col0, kk * 32);  // B[k][o] = kw[o][k]
    c = wmma_bf16(a, bb, c);
  }
  const int lane = threadIdx.x & 31;
  const int half = lane >> 4, cl = lane & 15;
  const int col  = col0 + cl;
  const float bias = kb[col];
#pragma unroll
  for (int r = 0; r < 8; ++r)
    Kb[(size_t)(row0 + half * 8 + r) * D_DIM + col] = (bf16_t)(c[r] + bias);
}

// ---------------------------------- qproj: Q[m] = codes_level @ W_level^T + b_level
__global__ __launch_bounds__(256) void hcg_qproj_kernel(
    const float* __restrict__ cat_c, const float* __restrict__ type_c,
    const float* __restrict__ var_c, const float* __restrict__ sp_c,
    const float* __restrict__ cat_w, const float* __restrict__ cat_b,
    const float* __restrict__ type_w, const float* __restrict__ type_b,
    const float* __restrict__ var_w, const float* __restrict__ var_b,
    const float* __restrict__ sp_w, const float* __restrict__ sp_b,
    bf16_t* __restrict__ Qb) {
  const int wid  = blockIdx.x * 8 + (threadIdx.x >> 5);
  const int m0   = (wid >> 4) * 16;
  const int col0 = (wid & 15) * 16;
  const float *codes, *W, *bias; int lm0;
  if (m0 < 16)       { codes = cat_c;  W = cat_w;  bias = cat_b;  lm0 = m0; }
  else if (m0 < 144) { codes = type_c; W = type_w; bias = type_b; lm0 = m0 - 16; }
  else if (m0 < 656) { codes = var_c;  W = var_w;  bias = var_b;  lm0 = m0 - 144; }
  else               { codes = sp_c;   W = sp_w;   bias = sp_b;   lm0 = m0 - 656; }
  v8f c;
#pragma unroll
  for (int r = 0; r < 8; ++r) c[r] = 0.0f;
#pragma unroll
  for (int kk = 0; kk < 8; ++kk) {
    v16bf a  = load_A(codes, D_DIM, lm0, kk * 32);
    v16bf bb = load_Bt(W, D_DIM, col0, kk * 32);
    c = wmma_bf16(a, bb, c);
  }
  const int lane = threadIdx.x & 31;
  const int half = lane >> 4, cl = lane & 15;
  const int col  = col0 + cl;
  const float bv = bias[col];
#pragma unroll
  for (int r = 0; r < 8; ++r)
    Qb[(size_t)(m0 + half * 8 + r) * D_DIM + col] = (bf16_t)(c[r] + bv);
}

// -------------------- attn: flash-style softmax(QK^T/16) @ X, plus ||H|| per row
__global__ __launch_bounds__(ATTN_WAVES * 32) void hcg_attn_kernel(
    const bf16_t* __restrict__ Qb, const bf16_t* __restrict__ Kb,
    const bf16_t* __restrict__ Xb, const unsigned char* __restrict__ mask,
    float* __restrict__ H, float* __restrict__ wraw) {
  __shared__ bf16_t sK[CHUNK][D_DIM + 8];              // K chunk, TDM-padded rows
  __shared__ bf16_t sXT[D_DIM][CHUNK + 2];             // X chunk transposed [d][token]
  __shared__ bf16_t sP[ATTN_WAVES][16][CHUNK + 2];     // per-wave P relayout buffer

  const int b    = blockIdx.y;
  const int w    = threadIdx.x >> 5;
  const int lane = threadIdx.x & 31;
  const int half = lane >> 4;
  const int cl   = lane & 15;
  const int m0   = (blockIdx.x * ATTN_WAVES + w) * 16;
  const int tid  = threadIdx.x;

  v16bf qf[8];
#pragma unroll
  for (int kk = 0; kk < 8; ++kk) qf[kk] = load_A(Qb, D_DIM, m0, kk * 32);

  v8f o[16];
#pragma unroll
  for (int dt = 0; dt < 16; ++dt)
#pragma unroll
    for (int r = 0; r < 8; ++r) o[dt][r] = 0.0f;

  float m_run[8], l_run[8];
#pragma unroll
  for (int r = 0; r < 8; ++r) { m_run[r] = -1e30f; l_run[r] = 0.0f; }

  const bf16_t*   Kbase = Kb + (size_t)b * N_TOK * D_DIM;
  const uint32_t* Xw = (const uint32_t*)(Xb + (size_t)b * N_TOK * D_DIM);
  const unsigned char* mk = mask + (size_t)b * N_TOK;
  const uint32_t sk_off = (uint32_t)(uintptr_t)&sK[0][0];  // LDS byte offset

  for (int n0 = 0; n0 < N_TOK; n0 += CHUNK) {
    __syncthreads();  // previous iteration's LDS reads done

    // Tensor Data Mover pulls the K tile; wave-uniform guard -> one issue per block
    if (w == 0) tdm_load_k_tile(Kbase + (size_t)n0 * D_DIM, sk_off);

    // meanwhile all waves stage X transposed (TDM cannot transpose)
    const uint32_t* xp = Xw + (size_t)n0 * (D_DIM / 2);
    for (int i = tid; i < CHUNK * D_DIM / 2; i += ATTN_WAVES * 32) {
      const int t  = i >> 7;             // token within chunk
      const int d2 = (i & 127) << 1;     // even d index
      union { uint32_t u; bf16_t h2[2]; } cv; cv.u = xp[i];
      sXT[d2][t]     = cv.h2[0];
      sXT[d2 + 1][t] = cv.h2[1];
    }
    if (n0 + CHUNK < N_TOK)              // global_prefetch_b8 next X chunk
      __builtin_prefetch(xp + CHUNK * (D_DIM / 2), 0, 1);

    if (w == 0) __builtin_amdgcn_s_wait_tensorcnt(0);
    __syncthreads();

    // S = Q @ K_chunk^T : two 16x16 tiles (tokens 0-15 and 16-31)
    v8f s0, s1;
#pragma unroll
    for (int r = 0; r < 8; ++r) { s0[r] = 0.0f; s1[r] = 0.0f; }
#pragma unroll
    for (int kk = 0; kk < 8; ++kk) {
      v16bf bk0 = load_Bt(&sK[0][0], D_DIM + 8, 0, kk * 32);
      s0 = wmma_bf16(qf[kk], bk0, s0);
      v16bf bk1 = load_Bt(&sK[0][0], D_DIM + 8, 16, kk * 32);
      s1 = wmma_bf16(qf[kk], bk1, s1);
    }

    const bool valid0 = mk[n0 + cl] != 0;
    const bool valid1 = mk[n0 + 16 + cl] != 0;
    float corr[8];
#pragma unroll
    for (int r = 0; r < 8; ++r) {
      float a0 = s0[r] * 0.0625f;                       // 1/sqrt(256)
      float a1 = s1[r] * 0.0625f;
      if (!valid0) a0 = -1.7e38f;                       // MASK_VAL
      if (!valid1) a1 = -1.7e38f;
      a0 = fminf(fmaxf(a0, -50.0f), 50.0f);             // clip
      a1 = fminf(fmaxf(a1, -50.0f), 50.0f);
      float mx = fmaxf(a0, a1);                         // row-max over 32 tokens
      mx = fmaxf(mx, __shfl_xor(mx, 1, 32));
      mx = fmaxf(mx, __shfl_xor(mx, 2, 32));
      mx = fmaxf(mx, __shfl_xor(mx, 4, 32));
      mx = fmaxf(mx, __shfl_xor(mx, 8, 32));
      const float c    = __expf(m_run[r] - fmaxf(m_run[r], mx));
      const float mnew = fmaxf(m_run[r], mx);
      const float p0   = __expf(a0 - mnew);
      const float p1   = __expf(a1 - mnew);
      float ps = p0 + p1;
      ps += __shfl_xor(ps, 1, 32);
      ps += __shfl_xor(ps, 2, 32);
      ps += __shfl_xor(ps, 4, 32);
      ps += __shfl_xor(ps, 8, 32);
      l_run[r] = l_run[r] * c + ps;
      m_run[r] = mnew;
      corr[r]  = c;
      sP[w][half * 8 + r][cl]      = (bf16_t)p0;        // D-layout -> LDS
      sP[w][half * 8 + r][16 + cl] = (bf16_t)p1;
    }

#pragma unroll
    for (int dt = 0; dt < 16; ++dt)
#pragma unroll
      for (int r = 0; r < 8; ++r) o[dt][r] *= corr[r];

    // O += P @ X_chunk  (P relayout from LDS as an A-fragment; wave-local DS in-order)
    v16bf pf = load_A(&sP[w][0][0], CHUNK + 2, 0, 0);
#pragma unroll
    for (int dt = 0; dt < 16; ++dt) {
      v16bf bx = load_Bt(&sXT[0][0], CHUNK + 2, dt * 16, 0);  // B[k][n]=X[tok k][d n]
      o[dt] = wmma_bf16(pf, bx, o[dt]);
    }
  }

  // normalize, write H and ||H||
  float inv[8], ssq[8];
#pragma unroll
  for (int r = 0; r < 8; ++r) { inv[r] = 1.0f / l_run[r]; ssq[r] = 0.0f; }
  float* hrow = H + ((size_t)b * M_TOT + m0) * D_DIM;
#pragma unroll
  for (int dt = 0; dt < 16; ++dt)
#pragma unroll
    for (int r = 0; r < 8; ++r) {
      const float v = o[dt][r] * inv[r];
      ssq[r] += v * v;
      hrow[(size_t)(half * 8 + r) * D_DIM + dt * 16 + cl] = v;
    }
#pragma unroll
  for (int r = 0; r < 8; ++r) {
    float s = ssq[r];
    s += __shfl_xor(s, 1, 32);
    s += __shfl_xor(s, 2, 32);
    s += __shfl_xor(s, 4, 32);
    s += __shfl_xor(s, 8, 32);
    if (cl == 0) wraw[(size_t)b * M_TOT + m0 + half * 8 + r] = sqrtf(s);
  }
}

// ------------------------------------------------ pos_pooled: masked mean over N
__global__ __launch_bounds__(256) void hcg_pospool_kernel(
    const float* __restrict__ positions, const unsigned char* __restrict__ mask,
    float* __restrict__ pp) {
  const int b = blockIdx.x, d = threadIdx.x;
  const float* p = positions + (size_t)b * N_TOK * D_DIM + d;
  const unsigned char* mk = mask + (size_t)b * N_TOK;
  float s = 0.0f, cnt = 0.0f;
  for (int n = 0; n < N_TOK; ++n) {
    const float mv = mk[n] ? 1.0f : 0.0f;
    s += p[(size_t)n * D_DIM] * mv;
    cnt += mv;
  }
  pp[b * D_DIM + d] = s / fmaxf(cnt, 1.0f);
}

// ------------------------------------------------------------ final combine
__device__ inline float blk_sum(float v, float* red) {
  const int t = threadIdx.x;
  red[t] = v; __syncthreads();
  for (int s = 128; s > 0; s >>= 1) {
    if (t < s) red[t] += red[t + s];
    __syncthreads();
  }
  const float r = red[0]; __syncthreads();
  return r;
}
__device__ inline float blk_max(float v, float* red) {
  const int t = threadIdx.x;
  red[t] = v; __syncthreads();
  for (int s = 128; s > 0; s >>= 1) {
    if (t < s) red[t] = fmaxf(red[t], red[t + s]);
    __syncthreads();
  }
  const float r = red[0]; __syncthreads();
  return r;
}

__global__ __launch_bounds__(256) void hcg_final_kernel(
    const float* __restrict__ wraw, const float* __restrict__ H,
    const float* __restrict__ pp, const float* __restrict__ log_tau,
    const float* __restrict__ g1_w, const float* __restrict__ g1_b,
    const float* __restrict__ g2_w, const float* __restrict__ g2_b,
    const float* __restrict__ out_w, const float* __restrict__ out_b,
    const float* __restrict__ ln_g, const float* __restrict__ ln_b,
    const float* __restrict__ level_w, float* __restrict__ out) {
  __shared__ float red[256];
  __shared__ float wl[M_TOT];   // sparsified weights, global m indexing
  __shared__ float zc[256], zt[256], zv[256], zs[256], hb[256], zb[256];
  const int b = blockIdx.x, t = threadIdx.x;
  const float tau = fminf(fmaxf(__expf(log_tau[0]) + 0.1f, 0.1f), 2.0f);
  const float* wr = wraw + (size_t)b * M_TOT;

  { // category (16), thr 0.1
    const float v = (t < 16) ? wr[t] / tau : -1e30f;
    const float mx = blk_max(v, red);
    const float e = (t < 16) ? __expf(v - mx) : 0.0f;
    const float s = blk_sum(e, red);
    const float wv = e / s;
    const float wsp = (wv > 0.1f) ? wv : 0.0f;
    const float ss = blk_sum(wsp, red);
    if (t < 16) wl[t] = wsp / (ss + 1e-8f);
    __syncthreads();
  }
  { // type (128) gated by category, thr 0.05
    const float v = (t < 128) ? (wr[16 + t] / tau) * wl[t >> 3] : -1e30f;
    const float mx = blk_max(v, red);
    const float e = (t < 128) ? __expf(v - mx) : 0.0f;
    const float s = blk_sum(e, red);
    const float wv = e / s;
    const float wsp = (wv > 0.05f) ? wv : 0.0f;
    const float ss = blk_sum(wsp, red);
    if (t < 128) wl[16 + t] = wsp / (ss + 1e-8f);
    __syncthreads();
  }
  { // variant (512) gated by type (repeat 4), thr 0.025 — 2 elems/thread
    float vv[2]; float lmax = -1e30f;
#pragma unroll
    for (int j = 0; j < 2; ++j) {
      const int i = t + j * 256;
      vv[j] = (wr[144 + i] / tau) * wl[16 + (i >> 2)];
      lmax = fmaxf(lmax, vv[j]);
    }
    const float mx = blk_max(lmax, red);
    float ee[2]; float ls = 0.0f;
#pragma unroll
    for (int j = 0; j < 2; ++j) { ee[j] = __expf(vv[j] - mx); ls += ee[j]; }
    const float s = blk_sum(ls, red);
    float wsp[2]; float lss = 0.0f;
#pragma unroll
    for (int j = 0; j < 2; ++j) {
      const float wv = ee[j] / s;
      wsp[j] = (wv > 0.025f) ? wv : 0.0f;
      lss += wsp[j];
    }
    const float ss = blk_sum(lss, red);
#pragma unroll
    for (int j = 0; j < 2; ++j) wl[144 + t + j * 256] = wsp[j] / (ss + 1e-8f);
    __syncthreads();
  }
  { // spatial (16), thr 0.1
    const float v = (t < 16) ? wr[656 + t] / tau : -1e30f;
    const float mx = blk_max(v, red);
    const float e = (t < 16) ? __expf(v - mx) : 0.0f;
    const float s = blk_sum(e, red);
    const float wv = e / s;
    const float wsp = (wv > 0.1f) ? wv : 0.0f;
    const float ss = blk_sum(wsp, red);
    if (t < 16) wl[656 + t] = wsp / (ss + 1e-8f);
    __syncthreads();
  }

  { // z_level[d] = sum_m w[m] * H[b][m][d]
    const float* Hb = H + (size_t)b * M_TOT * D_DIM;
    float a0 = 0.f, a1 = 0.f, a2 = 0.f, a3 = 0.f;
    for (int m = 0; m < 16; ++m)  a0 += wl[m]       * Hb[(size_t)m * D_DIM + t];
    for (int m = 0; m < 128; ++m) a1 += wl[16 + m]  * Hb[(size_t)(16 + m) * D_DIM + t];
    for (int m = 0; m < 512; ++m) a2 += wl[144 + m] * Hb[(size_t)(144 + m) * D_DIM + t];
    for (int m = 0; m < 16; ++m)  a3 += wl[656 + m] * Hb[(size_t)(656 + m) * D_DIM + t];
    zc[t] = a0; zt[t] = a1; zv[t] = a2; zs[t] = a3;
    __syncthreads();
  }

  // position gate: h = gelu(g1 @ [z_cat, pos_pooled]); pg = sigmoid(g2 @ h)
  float acc = g1_b[t];
  const float* g1r = g1_w + (size_t)t * 512;
  for (int i = 0; i < 256; ++i) acc += g1r[i] * zc[i];
  const float* ppb = pp + (size_t)b * D_DIM;
  for (int i = 0; i < 256; ++i) acc += g1r[256 + i] * ppb[i];
  hb[t] = 0.5f * acc * (1.0f + erff(acc * 0.70710678118f));
  __syncthreads();
  float acc2 = g2_b[t];
  const float* g2r = g2_w + (size_t)t * 256;
  for (int i = 0; i < 256; ++i) acc2 += g2r[i] * hb[i];
  const float pg = 1.0f / (1.0f + __expf(-acc2));

  // level-weight softmax + combine
  const float l0 = level_w[0], l1 = level_w[1], l2 = level_w[2], l3 = level_w[3];
  const float lm = fmaxf(fmaxf(l0, l1), fmaxf(l2, l3));
  const float e0 = __expf(l0 - lm), e1 = __expf(l1 - lm);
  const float e2 = __expf(l2 - lm), e3 = __expf(l3 - lm);
  const float es = e0 + e1 + e2 + e3;
  zb[t] = (e0 * (zc[t] * pg) + e1 * zt[t] + e2 * zv[t] + e3 * zs[t]) / es;
  __syncthreads();

  float y = out_b[t];
  const float* owr = out_w + (size_t)t * 256;
  for (int i = 0; i < 256; ++i) y += owr[i] * zb[i];

  const float mu  = blk_sum(y, red) * (1.0f / 256.0f);
  const float df  = y - mu;
  const float var = blk_sum(df * df, red) * (1.0f / 256.0f);
  out[(size_t)b * D_DIM + t] = df * rsqrtf(var + 1e-5f) * ln_g[t] + ln_b[t];
}

// ------------------------------------------------------------------ launch
extern "C" void kernel_launch(void* const* d_in, const int* in_sizes, int n_in,
                              void* d_out, int out_size, void* d_ws, size_t ws_size,
                              hipStream_t stream) {
  (void)in_sizes; (void)n_in; (void)out_size; (void)ws_size;
  const float* X          = (const float*)d_in[0];
  const float* positions  = (const float*)d_in[1];
  const float* cat_codes  = (const float*)d_in[2];
  const float* type_codes = (const float*)d_in[3];
  const float* var_codes  = (const float*)d_in[4];
  const float* sp_codes   = (const float*)d_in[5];
  const float* log_tau    = (const float*)d_in[6];
  const float* cat_w = (const float*)d_in[7];  const float* cat_b = (const float*)d_in[8];
  const float* type_w = (const float*)d_in[9]; const float* type_b = (const float*)d_in[10];
  const float* var_w = (const float*)d_in[11]; const float* var_b = (const float*)d_in[12];
  const float* sp_w = (const float*)d_in[13];  const float* sp_b = (const float*)d_in[14];
  const float* k_w = (const float*)d_in[15];   const float* k_b = (const float*)d_in[16];
  const float* g1_w = (const float*)d_in[17];  const float* g1_b = (const float*)d_in[18];
  const float* g2_w = (const float*)d_in[19];  const float* g2_b = (const float*)d_in[20];
  const float* out_w = (const float*)d_in[21]; const float* out_b = (const float*)d_in[22];
  const float* ln_g = (const float*)d_in[23];  const float* ln_b = (const float*)d_in[24];
  const float* level_w = (const float*)d_in[25];
  const unsigned char* mask = (const unsigned char*)d_in[26];

  char* ws = (char*)d_ws;
  size_t off = 0;
  bf16_t* Xb = (bf16_t*)(ws + off); off += (size_t)N_BATCH * N_TOK * D_DIM * 2;
  bf16_t* Kb = (bf16_t*)(ws + off); off += (size_t)N_BATCH * N_TOK * D_DIM * 2;
  bf16_t* Qb = (bf16_t*)(ws + off); off += (size_t)M_TOT * D_DIM * 2;
  float* H    = (float*)(ws + off); off += (size_t)N_BATCH * M_TOT * D_DIM * 4;
  float* wraw = (float*)(ws + off); off += (size_t)N_BATCH * M_TOT * 4;
  float* pp   = (float*)(ws + off); off += (size_t)N_BATCH * D_DIM * 4;

  const int n4 = N_BATCH * N_TOK * D_DIM / 4;
  hcg_cvt_kernel<<<(n4 + 255) / 256, 256, 0, stream>>>(X, Xb, n4);

  // 65536 token rows x 16 col-tiles, 8 waves/block
  hcg_kproj_kernel<<<(N_BATCH * N_TOK / 16) * (D_DIM / 16) / 8, 256, 0, stream>>>(
      X, k_w, k_b, Kb);

  // 42 row-tiles x 16 col-tiles = 672 waves -> 84 blocks
  hcg_qproj_kernel<<<(M_TOT / 16) * (D_DIM / 16) / 8, 256, 0, stream>>>(
      cat_codes, type_codes, var_codes, sp_codes,
      cat_w, cat_b, type_w, type_b, var_w, var_b, sp_w, sp_b, Qb);

  hcg_attn_kernel<<<dim3(M_TOT / 16 / ATTN_WAVES, N_BATCH), ATTN_WAVES * 32, 0,
                    stream>>>(Qb, Kb, Xb, mask, H, wraw);

  hcg_pospool_kernel<<<N_BATCH, 256, 0, stream>>>(positions, mask, pp);

  hcg_final_kernel<<<N_BATCH, 256, 0, stream>>>(
      wraw, H, pp, log_tau, g1_w, g1_b, g2_w, g2_b, out_w, out_b, ln_g, ln_b,
      level_w, (float*)d_out);
}
